// PrototypeMatch_9586367005335
// MI455X (gfx1250) — matrix operand order repaired and checked by reading, not compiled, not run on hardware
//
#include <hip/hip_runtime.h>

typedef __attribute__((ext_vector_type(16))) __bf16 v16bf;
typedef __attribute__((ext_vector_type(8)))  float  v8f;

union Frag16 { uint4 u4[2]; v16bf v; };   // 16 bf16 = 32 bytes = one WMMA 16-bit operand per lane

constexpr int Bb = 8, Ll = 8192, Cc = 256, Nn = 512, Kk = 16;
constexpr int ROWS_PER_BLOCK = 64;                    // 4 M-tiles of 16
constexpr int BLOCKS_PER_B   = Ll / ROWS_PER_BLOCK;   // 128

__device__ __forceinline__ unsigned short f32_to_bf16(float f) {
    union { float f; unsigned u; } x; x.f = f;
    unsigned r = x.u + 0x7FFFu + ((x.u >> 16) & 1u);   // round-to-nearest-even
    return (unsigned short)(r >> 16);
}

// ---- Phase 1: g[b][c] = mean_k global_token[b][k][c] ---------------------
__global__ void gmean_kernel(const float* __restrict__ gt, float* __restrict__ g) {
    int b = blockIdx.x, c = threadIdx.x;
    float s = 0.f;
#pragma unroll
    for (int k = 0; k < Kk; ++k) s += gt[(b * Kk + k) * Cc + c];
    g[b * Cc + c] = s * (1.0f / Kk);
}

// ---- Phase 2: prototypes f32 -> bf16, row-major [N][C] -------------------
__global__ void proto_cvt_kernel(const float* __restrict__ p, unsigned short* __restrict__ pb) {
    int i = blockIdx.x * blockDim.x + threadIdx.x;
    pb[i] = f32_to_bf16(p[i]);
}

// ---- Phase 3: 64-row tile: WMMA scores (4x B-reuse), argmax, distance ----
__global__ __launch_bounds__(256) void match_kernel(const float* __restrict__ q,
                                                    const float* __restrict__ protos_f32,
                                                    const unsigned short* __restrict__ protos_bf16,
                                                    const float* __restrict__ gmean,
                                                    float* __restrict__ out) {
    __shared__ __align__(16) unsigned short sA[ROWS_PER_BLOCK * Cc];  // bf16 (q+g), 32 KB
    __shared__ float sRedVal[8 * ROWS_PER_BLOCK];                     // per-wave bests
    __shared__ int   sRedIdx[8 * ROWS_PER_BLOCK];
    __shared__ int   sIdx[ROWS_PER_BLOCK];

    const int b    = blockIdx.x / BLOCKS_PER_B;
    const int row0 = (blockIdx.x % BLOCKS_PER_B) * ROWS_PER_BLOCK;
    const float* qbase = q + (size_t)(b * Ll + row0) * Cc;
    const float* gb    = gmean + b * Cc;
    const int tid = threadIdx.x;

    // Stage 64x256 bf16(q + g) for the WMMA A operand (coalesced, 64 KB read)
    for (int it = 0; it < ROWS_PER_BLOCK; ++it) {
        int f = it * 256 + tid;
        sA[f] = f32_to_bf16(qbase[f] + gb[f & (Cc - 1)]);
    }
    __syncthreads();

    const int wave = tid >> 5;
    const int lane = tid & 31;
    const int l15  = lane & 15;
    const bool lo  = (lane < 16);
    const int khalfA = lo ? 0 : 8;   // A 16x32: lanes0-15 K={0..7,16..23}; lanes16-31 K={8..15,24..31}
    const int khalfB = lo ? 0 : 8;   // B 32x16: lanes0-15 K={0..7,16..23}; lanes16-31 K={8..15,24..31}

    float bestv[4][8];
    int   besti[4][8];
#pragma unroll
    for (int mt = 0; mt < 4; ++mt)
#pragma unroll
        for (int r = 0; r < 8; ++r) { bestv[mt][r] = -3.4e38f; besti[mt][r] = 0x7FFFFFFF; }

    // Each wave: 4 N-tiles; each loaded B fragment feeds 4 WMMAs (one per M-tile)
#pragma unroll
    for (int jj = 0; jj < 4; ++jj) {
        const int nt = wave * 4 + jj;
        v8f acc[4] = {};
#pragma unroll
        for (int k0 = 0; k0 < Cc; k0 += 32) {
            Frag16 bfr;
            const unsigned short* bp = protos_bf16 + (size_t)(nt * 16 + l15) * Cc + k0 + khalfB;
            bfr.u4[0] = *(const uint4*)(bp);
            bfr.u4[1] = *(const uint4*)(bp + 16);
#pragma unroll
            for (int mt = 0; mt < 4; ++mt) {
                Frag16 af;
                const unsigned short* ap = &sA[(mt * 16 + l15) * Cc + k0 + khalfA];
                af.u4[0] = *(const uint4*)(ap);
                af.u4[1] = *(const uint4*)(ap + 16);
                acc[mt] = __builtin_amdgcn_wmma_f32_16x16x32_bf16(
                    false, af.v, false, bfr.v, (short)0, acc[mt], false, false);
            }
        }
        // Running argmax update: each lane owns one column n for this N-tile
        const int ncol = nt * 16 + l15;
#pragma unroll
        for (int mt = 0; mt < 4; ++mt)
#pragma unroll
            for (int r = 0; r < 8; ++r) {
                float v = acc[mt][r];
                if (v > bestv[mt][r]) { bestv[mt][r] = v; besti[mt][r] = ncol; }
            }
    }

    // Cross-lane argmax within each 16-lane half (C/D layout: N = lane&15)
#pragma unroll
    for (int mt = 0; mt < 4; ++mt)
#pragma unroll
        for (int r = 0; r < 8; ++r) {
            float bv = bestv[mt][r]; int bi = besti[mt][r];
            for (int m = 1; m < 16; m <<= 1) {
                float ov = __shfl_xor(bv, m, 32);
                int   oi = __shfl_xor(bi, m, 32);
                if (ov > bv || (ov == bv && oi < bi)) { bv = ov; bi = oi; }
            }
            if (l15 == 0) {
                int mrow = mt * 16 + r + (lo ? 0 : 8);  // VGPR r -> M=r (lanes<16) / M=8+r
                sRedVal[wave * ROWS_PER_BLOCK + mrow] = bv;
                sRedIdx[wave * ROWS_PER_BLOCK + mrow] = bi;
            }
        }
    __syncthreads();

    if (tid < ROWS_PER_BLOCK) {            // reduce across the 8 waves (disjoint N ranges)
        float bv = sRedVal[tid]; int bi = sRedIdx[tid];
#pragma unroll
        for (int w = 1; w < 8; ++w) {
            float ov = sRedVal[w * ROWS_PER_BLOCK + tid];
            int   oi = sRedIdx[w * ROWS_PER_BLOCK + tid];
            if (ov > bv || (ov == bv && oi < bi)) { bv = ov; bi = oi; }
        }
        sIdx[tid] = bi;
    }
    __syncthreads();

    // rd[m] = ||q[m] - proto[idx[m]]||^2 in f32; 16 threads per row, 4 passes
#pragma unroll
    for (int pass = 0; pass < 4; ++pass) {
        int m  = pass * 16 + (tid >> 4);
        int c0 = (tid & 15) * 16;
        const float* pr = protos_f32 + (size_t)sIdx[m] * Cc;
        const float* qr = qbase + (size_t)m * Cc;
        float s = 0.f;
#pragma unroll
        for (int i = 0; i < 16; ++i) {
            float d = qr[c0 + i] - pr[c0 + i];
            s += d * d;
        }
        for (int mm = 1; mm < 16; mm <<= 1) s += __shfl_xor(s, mm, 32);
        if ((tid & 15) == 0) out[(size_t)b * Ll + row0 + m] = s;
    }
}

extern "C" void kernel_launch(void* const* d_in, const int* in_sizes, int n_in,
                              void* d_out, int out_size, void* d_ws, size_t ws_size,
                              hipStream_t stream) {
    const float* q      = (const float*)d_in[0];   // [B,L,C]
    const float* protos = (const float*)d_in[1];   // [N,C]
    const float* gtok   = (const float*)d_in[2];   // [B,K,C]
    float* out = (float*)d_out;                    // [B,L]

    float*          g_ws = (float*)d_ws;                                   // 8*256 f32 = 8 KB
    unsigned short* p_bf = (unsigned short*)((char*)d_ws + Bb * Cc * 4);   // 512*256 bf16 = 256 KB

    gmean_kernel<<<dim3(Bb), dim3(Cc), 0, stream>>>(gtok, g_ws);
    proto_cvt_kernel<<<dim3((Nn * Cc) / 256), dim3(256), 0, stream>>>(protos, p_bf);
    match_kernel<<<dim3((Bb * Ll) / ROWS_PER_BLOCK), dim3(256), 0, stream>>>(
        q, protos, p_bf, g_ws, out);
}
